// TransformerBlock_2448131359431
// MI455X (gfx1250) — compile-verified, matrix operand order checked by reading, not run on hardware
//
#include <hip/hip_runtime.h>
#include <math.h>

// ---------------- problem constants ----------------
#define BB   2
#define SS   2048
#define DD   2048
#define HH   16
#define DHD  128
#define DFF  8192
#define MM   (BB * SS)   // 4096 rows

typedef __attribute__((ext_vector_type(16))) __bf16 bf16x16;
typedef __attribute__((ext_vector_type(8)))  float  f32x8;
typedef __attribute__((ext_vector_type(4)))  unsigned int u32x4;
typedef __attribute__((ext_vector_type(8)))  int i32x8;
typedef __attribute__((ext_vector_type(4)))  int i32x4;

union BF16Vec { bf16x16 v; unsigned short u[16]; };

static __device__ __forceinline__ unsigned short f32_to_bf16(float f) {
  union { float f; unsigned u; } w; w.f = f;
  unsigned r = w.u + 0x7FFFu + ((w.u >> 16) & 1u);  // round-nearest-even
  return (unsigned short)(r >> 16);
}

// ---- CDNA5 async global->LDS copy (ASYNCcnt-tracked, no VGPR round trip) ----
static __device__ __forceinline__ void async_g2l_b128(const void* lds_ptr,
                                                      const void* gptr) {
  unsigned laddr = (unsigned)(size_t)lds_ptr;  // low 32 bits = LDS byte offset
  unsigned long long gaddr = (unsigned long long)(size_t)gptr;
  asm volatile("global_load_async_to_lds_b128 %0, %1, off"
               :
               : "v"(laddr), "v"(gaddr)
               : "memory");
}
static __device__ __forceinline__ void wait_async0() {
  asm volatile("s_wait_asynccnt 0x0" ::: "memory");
}

// ---- CDNA5 Tensor Data Mover: one-shot 2D tile DMA global->LDS ----
#if __has_builtin(__builtin_amdgcn_tensor_load_to_lds)
#define HAVE_TDM 1
#endif

// Loads a [tile_d1 x tile_d0] tile (2-byte elements) whose rows are
// stride_d0 elements apart in global memory, packed row-major into LDS.
static __device__ __forceinline__ void tdm_load_tile_2d(
    const void* lds_ptr, const void* gptr,
    unsigned tile_d0, unsigned tile_d1,
    unsigned tensor_d0, unsigned tensor_d1, unsigned stride_d0) {
#ifdef HAVE_TDM
  unsigned long long ga = (unsigned long long)(size_t)gptr;
  unsigned lds = (unsigned)(size_t)lds_ptr;
  u32x4 g0;
  g0.x = 1u;                                              // count=1 (valid), user mode
  g0.y = lds;                                             // lds_addr
  g0.z = (unsigned)(ga & 0xFFFFFFFFu);                    // global_addr[31:0]
  g0.w = (unsigned)((ga >> 32) & 0x01FFFFFFu) | (2u << 30);  // addr[56:32] | type=2
  i32x8 g1;
  g1[0] = (int)(1u << 16);                                // data_size=1 -> 2 bytes
  g1[1] = (int)((tensor_d0 & 0xFFFFu) << 16);             // tensor_dim0[15:0]
  g1[2] = (int)((tensor_d0 >> 16) | ((tensor_d1 & 0xFFFFu) << 16));
  g1[3] = (int)((tensor_d1 >> 16) | ((tile_d0 & 0xFFFFu) << 16));
  g1[4] = (int)(tile_d1 & 0xFFFFu);                       // tile_dim1 | tile_dim2=0
  g1[5] = (int)stride_d0;                                 // dim0 stride [31:0]
  g1[6] = 0;                                              // stride hi | dim1 stride lo
  g1[7] = 0;
  i32x4 z4 = {0, 0, 0, 0};
  i32x8 z8 = {0, 0, 0, 0, 0, 0, 0, 0};
  __builtin_amdgcn_tensor_load_to_lds(g0, g1, z4, z4, z8, 0);
#else
  // fallback: per-lane async copies (16B per lane-chunk)
  const unsigned short* g = (const unsigned short*)gptr;
  unsigned short* l = (unsigned short*)lds_ptr;
  int lane = threadIdx.x & 31;
  for (unsigned r = lane; r < tile_d1; r += 32)
    for (unsigned c = 0; c < tile_d0; c += 8)
      async_g2l_b128(l + r * tile_d0 + c, g + (size_t)r * stride_d0 + c);
  wait_async0();
#endif
}
static __device__ __forceinline__ void wait_tensor0() {
#ifdef HAVE_TDM
  __builtin_amdgcn_s_wait_tensorcnt(0);
#endif
}

// ---------------- fp32 -> bf16 convert ----------------
__global__ void cvt_f32_to_bf16(const float* __restrict__ src,
                                unsigned short* __restrict__ dst, int n) {
  int i = blockIdx.x * blockDim.x + threadIdx.x;
  int stride = gridDim.x * blockDim.x;
  for (; i < n; i += stride) dst[i] = f32_to_bf16(src[i]);
}

// ---------------- layernorm (row = 2048) -> bf16 ----------------
__global__ __launch_bounds__(256) void layernorm_to_bf16(
    const float* __restrict__ x, const float* __restrict__ g,
    const float* __restrict__ b, unsigned short* __restrict__ out) {
  const int row = blockIdx.x;
  const float* xr = x + (size_t)row * DD;
  __shared__ float rs[256], rs2[256];
  float s = 0.f, s2 = 0.f;
  for (int i = threadIdx.x; i < DD; i += 256) {
    float v = xr[i]; s += v; s2 += v * v;
  }
  rs[threadIdx.x] = s; rs2[threadIdx.x] = s2;
  __syncthreads();
  for (int off = 128; off > 0; off >>= 1) {
    if (threadIdx.x < off) {
      rs[threadIdx.x]  += rs[threadIdx.x + off];
      rs2[threadIdx.x] += rs2[threadIdx.x + off];
    }
    __syncthreads();
  }
  float mean = rs[0] * (1.0f / DD);
  float var  = rs2[0] * (1.0f / DD) - mean * mean;
  float inv  = rsqrtf(var + 1e-5f);
  unsigned short* orow = out + (size_t)row * DD;
  for (int i = threadIdx.x; i < DD; i += 256)
    orow[i] = f32_to_bf16((xr[i] - mean) * inv * g[i] + b[i]);
}

// ---------------- tiled WMMA GEMM: C = A(bf16 MxK) * B(bf16 KxN) ----------------
// block tile 128x128, BK=32, 256 threads = 8 waves; wave -> 32x64 sub-tile.
// A tile staged by a single TDM descriptor DMA (wave 0); B tile transposed
// through registers so fragment reads stay contiguous ds_load_b128s.
template <bool GELU, bool RES, bool OUTBF>
__global__ __launch_bounds__(256) void gemm_bf16_wmma(
    const unsigned short* __restrict__ A, const unsigned short* __restrict__ Bw,
    const float* __restrict__ bias, const float* __restrict__ res,
    float* __restrict__ outF, unsigned short* __restrict__ outB,
    int M, int N, int K) {
  __shared__ __attribute__((aligned(32))) unsigned short sA[128 * 32];  // [m][k]
  __shared__ __attribute__((aligned(32))) unsigned short sB[128 * 32];  // [n][k] (transposed)

  const int tid  = threadIdx.x;
  const int lane = tid & 31;
  const int wave = tid >> 5;     // 0..7
  const int wm   = wave >> 1;    // 0..3  (rows)
  const int wn   = wave & 1;     // 0..1  (cols)
  const int hl   = lane >> 4;    // 0/1
  const int l16  = lane & 15;

  const int bm = blockIdx.y * 128;
  const int bn = blockIdx.x * 128;

  f32x8 acc[2][4];
#pragma unroll
  for (int mt = 0; mt < 2; ++mt)
#pragma unroll
    for (int nt = 0; nt < 4; ++nt)
      acc[mt][nt] = (f32x8){0.f, 0.f, 0.f, 0.f, 0.f, 0.f, 0.f, 0.f};

  for (int k0 = 0; k0 < K; k0 += 32) {
    // stage A tile (128 rows x 32 cols) with one TDM tile DMA from wave 0
    if (wave == 0) {
      tdm_load_tile_2d(sA, A + (size_t)bm * K + k0,
                       /*tile_d0=*/32, /*tile_d1=*/128,
                       /*tensor_d0=*/(unsigned)K, /*tensor_d1=*/(unsigned)M,
                       /*stride_d0=*/(unsigned)K);
    }
    // stage B tile (32x128) transposed into [n][k] through registers
#pragma unroll
    for (int i = 0; i < 2; ++i) {
      int c  = tid * 2 + i;
      int kr = c >> 4;
      int nc = (c & 15) * 8;
      uint4 d = *reinterpret_cast<const uint4*>(Bw + (size_t)(k0 + kr) * N + bn + nc);
      const unsigned short* dsv = reinterpret_cast<const unsigned short*>(&d);
#pragma unroll
      for (int j = 0; j < 8; ++j) sB[(nc + j) * 32 + kr] = dsv[j];
    }
    if (k0 + 32 < K) {  // prefetch next B tile slice into L2
      __builtin_prefetch(Bw + (size_t)(k0 + 32 + (tid >> 4)) * N + bn + (tid & 15) * 8, 0, 1);
    }
    wait_tensor0();
    __syncthreads();

    bf16x16 af[2];
#pragma unroll
    for (int mt = 0; mt < 2; ++mt) {
      int r = wm * 32 + mt * 16 + l16;
      af[mt] = *reinterpret_cast<const bf16x16*>(&sA[r * 32 + hl * 16]);
    }
    bf16x16 bf[4];
#pragma unroll
    for (int nt = 0; nt < 4; ++nt) {
      int n = wn * 64 + nt * 16 + l16;
      bf[nt] = *reinterpret_cast<const bf16x16*>(&sB[n * 32 + hl * 16]);
    }
#pragma unroll
    for (int mt = 0; mt < 2; ++mt)
#pragma unroll
      for (int nt = 0; nt < 4; ++nt)
        acc[mt][nt] = __builtin_amdgcn_wmma_f32_16x16x32_bf16(
            false, af[mt], false, bf[nt], (short)0, acc[mt][nt], false, false);
    __syncthreads();
  }

  // epilogue: bias (+residual) (+gelu), fp32 or bf16 out
#pragma unroll
  for (int mt = 0; mt < 2; ++mt) {
#pragma unroll
    for (int nt = 0; nt < 4; ++nt) {
#pragma unroll
      for (int r = 0; r < 8; ++r) {
        int row = bm + wm * 32 + mt * 16 + r + hl * 8;
        int col = bn + wn * 64 + nt * 16 + l16;
        float v = acc[mt][nt][r] + bias[col];
        if (RES)  v += res[(size_t)row * N + col];
        if (GELU) v = 0.5f * v * (1.0f + erff(v * 0.70710678118654752f));
        if (OUTBF) outB[(size_t)row * N + col] = f32_to_bf16(v);
        else       outF[(size_t)row * N + col] = v;
      }
    }
  }
}

// ---------------- flash-attention (causal), one wave per (b,h,16-query tile) ----------------
__global__ __launch_bounds__(32) void attention_wmma(
    const unsigned short* __restrict__ q, const unsigned short* __restrict__ k,
    const unsigned short* __restrict__ v, unsigned short* __restrict__ o) {
  __shared__ __attribute__((aligned(32))) unsigned short sP[16 * 32];   // P tile (m x key)
  __shared__ __attribute__((aligned(32))) unsigned short sV[32 * 128];  // V block [key][dh]

  const int lane = threadIdx.x & 31;
  const int hl   = lane >> 4;
  const int l16  = lane & 15;
  const int q0   = blockIdx.x * 16;
  const int h    = blockIdx.y;
  const int b    = blockIdx.z;

  // Q fragments: 16x128 split into 4 K-steps of 32
  bf16x16 qf[4];
  const size_t baseQ = ((size_t)b * SS + q0) * DD + (size_t)h * DHD;
#pragma unroll
  for (int kk = 0; kk < 4; ++kk)
    qf[kk] = *reinterpret_cast<const bf16x16*>(
        q + baseQ + (size_t)l16 * DD + kk * 32 + hl * 16);

  f32x8 oacc[8];
#pragma unroll
  for (int t = 0; t < 8; ++t)
    oacc[t] = (f32x8){0.f, 0.f, 0.f, 0.f, 0.f, 0.f, 0.f, 0.f};
  float mrow[8], lrow[8];
#pragma unroll
  for (int r = 0; r < 8; ++r) { mrow[r] = -3.0e38f; lrow[r] = 0.f; }

  const float scale = 0.08838834764831845f;  // 1/sqrt(128)
  const int nkb = (q0 + 16 + 31) / 32;       // causal: keys <= q0+15

  for (int kb = 0; kb < nkb; ++kb) {
    const int key0 = kb * 32;
    // stage V block (32 keys x 128 dh) into LDS via async copies
    {
      const unsigned short* vr = v + ((size_t)b * SS + key0 + lane) * DD + (size_t)h * DHD;
#pragma unroll
      for (int j = 0; j < 16; ++j)
        async_g2l_b128(&sV[lane * 128 + j * 8], vr + j * 8);
    }
    // load all 8 K fragments up front (lets loads overlap WMMA issue)
    bf16x16 kf[2][4];
#pragma unroll
    for (int nt = 0; nt < 2; ++nt) {
      const int key = key0 + nt * 16 + l16;
      const size_t baseK = ((size_t)b * SS + key) * DD + (size_t)h * DHD;
#pragma unroll
      for (int kk = 0; kk < 4; ++kk)
        kf[nt][kk] = *reinterpret_cast<const bf16x16*>(k + baseK + kk * 32 + hl * 16);
    }
    // scores: 16x32 = two 16x16 WMMA tiles, 4 K-steps each over DH
    f32x8 sc[2];
#pragma unroll
    for (int nt = 0; nt < 2; ++nt) {
      sc[nt] = (f32x8){0.f, 0.f, 0.f, 0.f, 0.f, 0.f, 0.f, 0.f};
#pragma unroll
      for (int kk = 0; kk < 4; ++kk)
        sc[nt] = __builtin_amdgcn_wmma_f32_16x16x32_bf16(
            false, qf[kk], false, kf[nt][kk], (short)0, sc[nt], false, false);
    }
    // causal mask + online softmax (row spread across 16 lanes of a half-wave)
    float p0s[8], p1s[8];
#pragma unroll
    for (int r = 0; r < 8; ++r) {
      const int row = r + hl * 8;
      const int qi  = q0 + row;
      float s0 = sc[0][r] * scale; if (key0 + l16 > qi)      s0 = -3.0e38f;
      float s1 = sc[1][r] * scale; if (key0 + 16 + l16 > qi) s1 = -3.0e38f;
      float sm = fmaxf(s0, s1);
#pragma unroll
      for (int off = 8; off > 0; off >>= 1) sm = fmaxf(sm, __shfl_xor(sm, off, 32));
      float mn = fmaxf(mrow[r], sm);
      float p0 = __expf(s0 - mn);
      float p1 = __expf(s1 - mn);
      float ps = p0 + p1;
#pragma unroll
      for (int off = 8; off > 0; off >>= 1) ps += __shfl_xor(ps, off, 32);
      float corr = __expf(mrow[r] - mn);
      lrow[r] = lrow[r] * corr + ps;
      mrow[r] = mn;
      p0s[r] = p0; p1s[r] = p1;
#pragma unroll
      for (int t = 0; t < 8; ++t) oacc[t][r] *= corr;
    }
    // re-layout P (C-layout -> A-fragment) through LDS
#pragma unroll
    for (int r = 0; r < 8; ++r) {
      const int row = r + hl * 8;
      sP[row * 32 + l16]      = f32_to_bf16(p0s[r]);
      sP[row * 32 + 16 + l16] = f32_to_bf16(p1s[r]);
    }
    wait_async0();   // V block resident in LDS
    __syncthreads();
    bf16x16 pf = *reinterpret_cast<const bf16x16*>(&sP[l16 * 32 + hl * 16]);
    // O += P(16x32) * V(32x128): 8 N-tiles
#pragma unroll
    for (int t = 0; t < 8; ++t) {
      BF16Vec vb;
#pragma unroll
      for (int j = 0; j < 16; ++j)
        vb.u[j] = sV[(hl * 16 + j) * 128 + t * 16 + l16];
      oacc[t] = __builtin_amdgcn_wmma_f32_16x16x32_bf16(
          false, pf, false, vb.v, (short)0, oacc[t], false, false);
    }
    __syncthreads();
  }
  // normalize + write bf16 [B,S,D] at head column block
#pragma unroll
  for (int t = 0; t < 8; ++t)
#pragma unroll
    for (int r = 0; r < 8; ++r) {
      const int row = r + hl * 8;
      float val = oacc[t][r] / lrow[r];
      o[((size_t)b * SS + q0 + row) * DD + (size_t)h * DHD + t * 16 + l16] =
          f32_to_bf16(val);
    }
}

// ---------------- host orchestration ----------------
extern "C" void kernel_launch(void* const* d_in, const int* in_sizes, int n_in,
                              void* d_out, int out_size, void* d_ws, size_t ws_size,
                              hipStream_t stream) {
  (void)in_sizes; (void)n_in; (void)out_size; (void)ws_size;
  const float* x     = (const float*)d_in[0];
  const float* ln1_g = (const float*)d_in[2];
  const float* ln1_b = (const float*)d_in[3];
  const float* wq = (const float*)d_in[4];  const float* bq = (const float*)d_in[5];
  const float* wk = (const float*)d_in[6];  const float* bk = (const float*)d_in[7];
  const float* wv = (const float*)d_in[8];  const float* bv = (const float*)d_in[9];
  const float* wo = (const float*)d_in[10]; const float* bo = (const float*)d_in[11];
  const float* ln2_g = (const float*)d_in[12];
  const float* ln2_b = (const float*)d_in[13];
  const float* w_in  = (const float*)d_in[14]; const float* b_in  = (const float*)d_in[15];
  const float* w_out = (const float*)d_in[16]; const float* b_out = (const float*)d_in[17];
  float* out = (float*)d_out;

  char* ws = (char*)d_ws;
  size_t off = 0;
  auto alloc = [&](size_t bytes) -> void* {
    void* p = ws + off; off += (bytes + 255) & ~(size_t)255; return p;
  };
  unsigned short* wq_b   = (unsigned short*)alloc((size_t)DD * DD * 2);
  unsigned short* wk_b   = (unsigned short*)alloc((size_t)DD * DD * 2);
  unsigned short* wv_b   = (unsigned short*)alloc((size_t)DD * DD * 2);
  unsigned short* wo_b   = (unsigned short*)alloc((size_t)DD * DD * 2);
  unsigned short* win_b  = (unsigned short*)alloc((size_t)DD * DFF * 2);
  unsigned short* wout_b = (unsigned short*)alloc((size_t)DFF * DD * 2);
  unsigned short* xn     = (unsigned short*)alloc((size_t)MM * DD * 2);
  unsigned short* qb     = (unsigned short*)alloc((size_t)MM * DD * 2);
  unsigned short* kb     = (unsigned short*)alloc((size_t)MM * DD * 2);
  unsigned short* vb     = (unsigned short*)alloc((size_t)MM * DD * 2);
  unsigned short* attnb  = (unsigned short*)alloc((size_t)MM * DD * 2);
  float*          x1     = (float*)alloc((size_t)MM * DD * 4);
  unsigned short* xn2    = (unsigned short*)alloc((size_t)MM * DD * 2);
  unsigned short* hb     = (unsigned short*)alloc((size_t)MM * DFF * 2);

  // 1) weights -> bf16
  cvt_f32_to_bf16<<<2048, 256, 0, stream>>>(wq, wq_b, DD * DD);
  cvt_f32_to_bf16<<<2048, 256, 0, stream>>>(wk, wk_b, DD * DD);
  cvt_f32_to_bf16<<<2048, 256, 0, stream>>>(wv, wv_b, DD * DD);
  cvt_f32_to_bf16<<<2048, 256, 0, stream>>>(wo, wo_b, DD * DD);
  cvt_f32_to_bf16<<<4096, 256, 0, stream>>>(w_in, win_b, DD * DFF);
  cvt_f32_to_bf16<<<4096, 256, 0, stream>>>(w_out, wout_b, DFF * DD);

  // 2) LN1
  layernorm_to_bf16<<<MM, 256, 0, stream>>>(x, ln1_g, ln1_b, xn);

  // 3) QKV projections (bf16 out, fused bias)
  dim3 gD(DD / 128, MM / 128);
  gemm_bf16_wmma<false, false, true><<<gD, 256, 0, stream>>>(
      xn, wq_b, bq, nullptr, nullptr, qb, MM, DD, DD);
  gemm_bf16_wmma<false, false, true><<<gD, 256, 0, stream>>>(
      xn, wk_b, bk, nullptr, nullptr, kb, MM, DD, DD);
  gemm_bf16_wmma<false, false, true><<<gD, 256, 0, stream>>>(
      xn, wv_b, bv, nullptr, nullptr, vb, MM, DD, DD);

  // 4) causal flash attention
  attention_wmma<<<dim3(SS / 16, HH, BB), 32, 0, stream>>>(qb, kb, vb, attnb);

  // 5) output projection + residual (fp32 out)
  gemm_bf16_wmma<false, true, false><<<gD, 256, 0, stream>>>(
      attnb, wo_b, bo, x, x1, nullptr, MM, DD, DD);

  // 6) LN2
  layernorm_to_bf16<<<MM, 256, 0, stream>>>(x1, ln2_g, ln2_b, xn2);

  // 7) MLP up + exact GELU (bf16 out)
  dim3 gF(DFF / 128, MM / 128);
  gemm_bf16_wmma<true, false, true><<<gF, 256, 0, stream>>>(
      xn2, win_b, b_in, nullptr, nullptr, hb, MM, DFF, DD);

  // 8) MLP down + residual -> final fp32 output
  gemm_bf16_wmma<false, true, false><<<gD, 256, 0, stream>>>(
      hb, wout_b, b_out, x1, out, nullptr, MM, DD, DFF);
}
